// RelEdgeUpdate_42528766165729
// MI455X (gfx1250) — compile-verified, matrix operand order checked by reading, not run on hardware
//
#include <hip/hip_runtime.h>

// ---------------------------------------------------------------------------
// GAT edge update, MI455X (gfx1250, wave32).
//   z1 = src_h[src_idx] @ W + b            (600000 x 128, K = 128)
//   a  = z1 @ Wa_z + self_h[dst_idx] @ Wa_s + attn_b ;  e = leaky_relu(a)
// HBM-bound (~310 MB of z1/e writes; gathers are L2-resident). GEMM runs on
// v_wmma_f32_16x16x32_f16 with f16 hi/lo error compensation (3 WMMAs per
// tile-K-step) so accuracy stays ~f32 while compute stays under the roofline.
// Stage-2 attention is fused to avoid re-reading z1 from HBM.
// ---------------------------------------------------------------------------

typedef __attribute__((ext_vector_type(16))) _Float16 v16h;
typedef __attribute__((ext_vector_type(8)))  float    v8f;

#define NEG_SLOPE 0.01f

union Frag16 {            // one WMMA A/B operand: 16 halves = 2 x 16B chunks
    v16h  h;
    uint4 u[2];
};

union Half8 {             // 8 halves <-> one 16B chunk
    _Float16 e[8];
    uint4    u;
};

// ---- Pre-convert W (128x128 f32, row-major [k][n]) into transposed f16
// ---- hi/lo pairs Wth/Wtl laid out [n][k] so B-fragments are contiguous.
__global__ __launch_bounds__(256) void wt_convert_kernel(
    const float* __restrict__ W,
    _Float16* __restrict__ Wth,
    _Float16* __restrict__ Wtl)
{
    const int i = blockIdx.x * 256 + threadIdx.x;   // 16384 elements
    const int n = i >> 7;
    const int k = i & 127;
    const float w   = W[k * 128 + n];
    const _Float16 hi = (_Float16)w;
    const _Float16 lo = (_Float16)(w - (float)hi);
    Wth[i] = hi;            // Wth[n*128 + k]
    Wtl[i] = lo;
}

// ---- Fused gather + GEMM (WMMA) + attention + leaky-relu.
// Block: 256 threads = 8 waves. Tile: 16 edges x 128 output features.
// Wave w computes the 16x16 tile for features [16w, 16w+16).
__global__ __launch_bounds__(256) void gat_fused_kernel(
    const float* __restrict__ src_h,
    const float* __restrict__ self_h,
    const int*   __restrict__ src_idx,
    const int*   __restrict__ dst_idx,
    const _Float16* __restrict__ Wth,
    const _Float16* __restrict__ Wtl,
    const float* __restrict__ bias_v,
    const float* __restrict__ attn_W,   // [256]: Wa_z = [0..127], Wa_s = [128..255]
    const float* __restrict__ attn_b,
    float* __restrict__ z1,
    float* __restrict__ e_out)
{
    __shared__ __align__(16) _Float16 A_hi[16][128];  // 4 KB
    __shared__ __align__(16) _Float16 A_lo[16][128];  // 4 KB
    __shared__ float a_acc[16];

    const int  t     = threadIdx.x;
    const int  lane  = t & 31;
    const int  wave  = t >> 5;
    const long edge0 = (long)blockIdx.x * 16;

    // ---- Stage gathered src rows into LDS, split f32 -> f16 hi + lo -------
    {
        const int  r   = t >> 4;            // edge row 0..15
        const int  c   = (t & 15) * 8;      // 8 consecutive features
        const long row = (long)src_idx[edge0 + r] * 128;
        const float4 x0 = *(const float4*)(src_h + row + c);
        const float4 x1 = *(const float4*)(src_h + row + c + 4);
        const float xs[8] = {x0.x, x0.y, x0.z, x0.w, x1.x, x1.y, x1.z, x1.w};
        Half8 hs, ls;
#pragma unroll
        for (int i = 0; i < 8; ++i) {
            hs.e[i] = (_Float16)xs[i];
            ls.e[i] = (_Float16)(xs[i] - (float)hs.e[i]);
        }
        *(uint4*)&A_hi[r][c] = hs.u;
        *(uint4*)&A_lo[r][c] = ls.u;
    }

    // ---- Self term: a_acc[e] = attn_b + dot(self_h[dst[e]], Wa_s) ---------
    // Wave w handles edges 2w and 2w+1; full 32-lane shuffle reduction.
    {
        const float4 was = *(const float4*)(attn_W + 128 + lane * 4);
#pragma unroll
        for (int e2 = 0; e2 < 2; ++e2) {
            const int  e    = wave * 2 + e2;
            const long drow = (long)dst_idx[edge0 + e] * 128;
            const float4 v  = *(const float4*)(self_h + drow + lane * 4);
            float p = v.x * was.x + v.y * was.y + v.z * was.z + v.w * was.w;
#pragma unroll
            for (int off = 16; off > 0; off >>= 1)
                p += __shfl_xor(p, off, 32);
            if (lane == 0) a_acc[e] = p + attn_b[0];
        }
    }
    __syncthreads();

    // ---- WMMA GEMM: C(16x16) = A(16x128) x B(128x16), f16 split-compensated
    // A layout (16-bit, 16x32 per step): lanes 0-15 hold M=lane, K {0-7,16-23};
    // lanes 16-31 hold M=lane-16, K {8-15,24-31}. B mirrors with N per lane.
    const int nl     = lane & 15;             // N within tile / A row
    const int khalf  = (lane >> 4) * 8;       // K sub-offset for this lane
    const int n_glob = wave * 16 + nl;        // global output feature

    const _Float16* bh_row = Wth + n_glob * 128;
    const _Float16* bl_row = Wtl + n_glob * 128;

    v8f c = {};
#pragma unroll
    for (int k0 = 0; k0 < 128; k0 += 32) {
        Frag16 ah, al, bh, bl;
        ah.u[0] = *(const uint4*)&A_hi[nl][k0 + khalf];
        ah.u[1] = *(const uint4*)&A_hi[nl][k0 + khalf + 16];
        al.u[0] = *(const uint4*)&A_lo[nl][k0 + khalf];
        al.u[1] = *(const uint4*)&A_lo[nl][k0 + khalf + 16];
        bh.u[0] = *(const uint4*)(bh_row + k0 + khalf);
        bh.u[1] = *(const uint4*)(bh_row + k0 + khalf + 16);
        bl.u[0] = *(const uint4*)(bl_row + k0 + khalf);
        bl.u[1] = *(const uint4*)(bl_row + k0 + khalf + 16);
        // z ~= ah*bh + al*bh + ah*bl   (al*bl ~ 2^-22, dropped)
        c = __builtin_amdgcn_wmma_f32_16x16x32_f16(false, ah.h, false, bh.h,
                                                   (short)0, c, false, false);
        c = __builtin_amdgcn_wmma_f32_16x16x32_f16(false, al.h, false, bh.h,
                                                   (short)0, c, false, false);
        c = __builtin_amdgcn_wmma_f32_16x16x32_f16(false, ah.h, false, bl.h,
                                                   (short)0, c, false, false);
    }

    // ---- Bias, z1 store, and fused z1 @ Wa_z partials ---------------------
    // C layout: VGPR v, lanes 0-15 -> (M=v, N=lane); lanes 16-31 -> (M=v+8).
    const int   mbase = (lane >> 4) * 8;
    const float bv    = bias_v[n_glob];
    const float wz    = attn_W[n_glob];
    float p[8];
#pragma unroll
    for (int v = 0; v < 8; ++v) {
        const float zv = c[v] + bv;
        p[v] = zv * wz;
        z1[(edge0 + mbase + v) * 128 + n_glob] = zv;
    }

    // Reduce partials over the 16 N-lanes sharing the same M rows.
#pragma unroll
    for (int v = 0; v < 8; ++v)
#pragma unroll
        for (int off = 1; off < 16; off <<= 1)
            p[v] += __shfl_xor(p[v], off, 16);

    if (nl == 0) {
#pragma unroll
        for (int v = 0; v < 8; ++v)
            atomicAdd(&a_acc[mbase + v], p[v]);   // ds_add_f32
    }
    __syncthreads();

    // ---- Leaky ReLU + e store --------------------------------------------
    if (t < 16) {
        const float a = a_acc[t];
        e_out[edge0 + t] = (a >= 0.0f) ? a : NEG_SLOPE * a;
    }
}

extern "C" void kernel_launch(void* const* d_in, const int* in_sizes, int n_in,
                              void* d_out, int out_size, void* d_ws, size_t ws_size,
                              hipStream_t stream)
{
    const float* src_h  = (const float*)d_in[0];
    const float* self_h = (const float*)d_in[1];
    const int*   src_ix = (const int*)d_in[2];
    const int*   dst_ix = (const int*)d_in[3];
    const float* W      = (const float*)d_in[4];
    const float* b      = (const float*)d_in[5];
    const float* attn_W = (const float*)d_in[6];
    const float* attn_b = (const float*)d_in[7];

    const int n_edges = in_sizes[2];            // 600000, multiple of 16
    float* z1 = (float*)d_out;                  // [n_edges * 128]
    float* e  = z1 + (size_t)n_edges * 128;     // [n_edges]

    _Float16* Wth = (_Float16*)d_ws;            // 32 KB
    _Float16* Wtl = Wth + 128 * 128;            // 32 KB  (needs 64 KB of ws)

    wt_convert_kernel<<<64, 256, 0, stream>>>(W, Wth, Wtl);
    gat_fused_kernel<<<n_edges / 16, 256, 0, stream>>>(
        src_h, self_h, src_ix, dst_ix, Wth, Wtl, b, attn_W, attn_b, z1, e);
}
